// GNNModel_58274116272680
// MI455X (gfx1250) — matrix operand order repaired and checked by reading, not compiled, not run on hardware
//
#include <hip/hip_runtime.h>
#include <hip/hip_bf16.h>

#define D_IN   768
#define D_OUT  64
#define MTILE  32                    // nodes per block in the GEMM kernel
#define KSTEP  32                    // bf16 WMMA K per step
#define NKS    (D_IN / KSTEP)        // 24 k-steps
#define LDS_STRIDE (D_IN + 8)        // bf16 elems; 388 dwords/row -> stride%64banks = 4
#define NT_TOTAL 16                  // 256 output cols / 16

typedef __attribute__((ext_vector_type(16))) __bf16 v16bf;
typedef __attribute__((ext_vector_type(8)))  float  v8f;

__device__ __forceinline__ __bf16 f2bf(float f) { return (__bf16)f; }

// monotonic float <-> orderable-uint mapping for atomicMax-based segment max
__device__ __forceinline__ unsigned int ford(float f) {
    unsigned int u = __float_as_uint(f);
    return (u & 0x80000000u) ? ~u : (u | 0x80000000u);
}
__device__ __forceinline__ float funord(unsigned int u) {
    return __uint_as_float((u & 0x80000000u) ? (u ^ 0x80000000u) : ~u);
}

// ---------------------------------------------------------------- init
__global__ __launch_bounds__(256) void init_kernel(unsigned int* __restrict__ segmax,
                                                   float* __restrict__ denom, int n) {
    int i = blockIdx.x * 256 + threadIdx.x;
    if (i < n) { segmax[i] = 0u; denom[i] = 0.0f; }
}

// ---------------------------------------------------------------- pack weights -> bf16 B-fragments
// Wp layout: [(nt, ks, lane)] * 16 bf16. Lane l of the WMMA B operand (32K x 16N tile)
// holds N = l&15, K = ks*32 + 16*(l>>4) .. +15 (contiguous).
__global__ __launch_bounds__(256) void pack_w_kernel(
    const float* __restrict__ Wq, const float* __restrict__ Wk,
    const float* __restrict__ Wv, const float* __restrict__ Ws,
    __bf16* __restrict__ Wp)
{
    const int idx = blockIdx.x * 256 + threadIdx.x;        // 0 .. 16*24*32-1
    if (idx >= NT_TOTAL * NKS * 32) return;
    const int lane = idx & 31;
    const int ks   = (idx >> 5) % NKS;
    const int nt   = idx / (NKS * 32);
    const int sel  = nt >> 2;
    const float* W = (sel == 0) ? Wq : (sel == 1) ? Wk : (sel == 2) ? Wv : Ws;
    const int ccol = (nt & 3) * 16 + (lane & 15);          // row of W == col of B
    const int k0   = ks * KSTEP + 16 * (lane >> 4);
    const float* src = W + (size_t)ccol * D_IN + k0;
    __bf16* dst = Wp + (size_t)idx * 16;
    #pragma unroll
    for (int i = 0; i < 16; i += 4) {
        const float4 f = *(const float4*)(src + i);
        dst[i + 0] = f2bf(f.x); dst[i + 1] = f2bf(f.y);
        dst[i + 2] = f2bf(f.z); dst[i + 3] = f2bf(f.w);
    }
}

// ---------------------------------------------------------------- fused QKVS GEMM (WMMA bf16)
// Block: 256 threads (8 waves), 32 nodes (2 M-tiles). Wave w computes a 2Mx2N grid of
// 16x16 tiles: N-tiles {2w, 2w+1} (always within one weight matrix: sel = w>>1),
// both M-tiles. Each A and B fragment feeds two WMMAs.
__global__ __launch_bounds__(256) void gemm_qkvs_kernel(
    const float* __restrict__ x, const __bf16* __restrict__ Wp,
    const float* __restrict__ bq, const float* __restrict__ bk,
    const float* __restrict__ bv, const float* __restrict__ bs,
    float* __restrict__ Q, float* __restrict__ Kmat, float* __restrict__ V,
    float* __restrict__ Sout, int nNodes)
{
    __shared__ __bf16 xs[MTILE * LDS_STRIDE];   // 32 x 776 bf16 = 48.5 KB

    const int tid  = threadIdx.x;
    const int lane = tid & 31;
    const int wave = tid >> 5;
    const int l15  = lane & 15;
    const int lhi  = lane >> 4;
    const int nodeBase = blockIdx.x * MTILE;

    // packed B base for this wave: nt0 = 2*wave, this lane
    const __bf16* wp = Wp + ((size_t)(wave * 2) * NKS * 32 + lane) * 16;
    // warm the L2/L0 path for the first B fragments (global_prefetch_b8)
    __builtin_prefetch(wp, 0, 0);
    __builtin_prefetch(wp + (size_t)NKS * 512, 0, 0);

    // Stage x tile (fp32 -> bf16) into LDS; shift-only indexing:
    // thread handles row tid>>3, float4-chunks (tid&7) + 8*it
    {
        int row = nodeBase + (tid >> 3);
        if (row >= nNodes) row = nNodes - 1;    // clamp; stores are guarded later
        const float* xrow = x + (size_t)row * D_IN;
        __bf16* lrow = &xs[(tid >> 3) * LDS_STRIDE];
        #pragma unroll
        for (int it = 0; it < D_IN / 32; ++it) {   // 24 iterations
            const int c4 = ((tid & 7) + it * 8) * 4;
            const float4 f = *(const float4*)(xrow + c4);
            __bf16* p = lrow + c4;
            p[0] = f2bf(f.x); p[1] = f2bf(f.y); p[2] = f2bf(f.z); p[3] = f2bf(f.w);
        }
    }
    __syncthreads();

    const int sel = wave >> 1;        // which matrix this wave produces (uniform)
    float* dst        = (sel == 0) ? Q  : (sel == 1) ? Kmat : (sel == 2) ? V  : Sout;
    const float* bptr = (sel == 0) ? bq : (sel == 1) ? bk   : (sel == 2) ? bv : bs;

    v8f acc[2][2];
    #pragma unroll
    for (int mi = 0; mi < 2; ++mi)
        #pragma unroll
        for (int j = 0; j < 2; ++j)
            #pragma unroll
            for (int r = 0; r < 8; ++r) acc[mi][j][r] = 0.0f;

    const int arow0 = l15 * LDS_STRIDE + 16 * lhi;             // M-tile 0
    const int arow1 = (16 + l15) * LDS_STRIDE + 16 * lhi;      // M-tile 1

    #pragma unroll 4
    for (int ks = 0; ks < NKS; ++ks) {
        // A fragments from LDS (16-bit A 16x32 layout): two b128 reads each
        v16bf a0, a1;
        {
            const __bf16* p0 = &xs[arow0 + ks * KSTEP];
            const __bf16* p1 = &xs[arow1 + ks * KSTEP];
            #pragma unroll
            for (int i = 0; i < 8; ++i) {
                a0[i] = p0[i]; a0[i + 8] = p0[i + 8];
                a1[i] = p1[i]; a1[i + 8] = p1[i + 8];
            }
        }
        const v16bf b0 = *(const v16bf*)(wp + (size_t)ks * 512);
        const v16bf b1 = *(const v16bf*)(wp + (size_t)(NKS + ks) * 512);

        acc[0][0] = __builtin_amdgcn_wmma_f32_16x16x32_bf16(
            false, a0, false, b0, (short)0, acc[0][0], false, false);
        acc[0][1] = __builtin_amdgcn_wmma_f32_16x16x32_bf16(
            false, a0, false, b1, (short)0, acc[0][1], false, false);
        acc[1][0] = __builtin_amdgcn_wmma_f32_16x16x32_bf16(
            false, a1, false, b0, (short)0, acc[1][0], false, false);
        acc[1][1] = __builtin_amdgcn_wmma_f32_16x16x32_bf16(
            false, a1, false, b1, (short)0, acc[1][1], false, false);
    }

    // C/D layout: VGPR r -> M = r + 8*lhi, N = l15
    #pragma unroll
    for (int mi = 0; mi < 2; ++mi) {
        #pragma unroll
        for (int j = 0; j < 2; ++j) {
            const int ccol = ((wave * 2 + j) & 3) * 16 + l15;
            const float bias = bptr[ccol];
            #pragma unroll
            for (int r = 0; r < 8; ++r) {
                const int row = nodeBase + mi * 16 + r + 8 * lhi;
                if (row < nNodes)
                    dst[(size_t)row * D_OUT + ccol] = acc[mi][j][r] + bias;
            }
        }
    }
}

// ---------------------------------------------------------------- per-edge score + segment max
__global__ __launch_bounds__(256) void score_kernel(
    const int* __restrict__ srcIdx, const int* __restrict__ dstIdx,
    const float* __restrict__ Q, const float* __restrict__ K,
    float* __restrict__ score, unsigned int* __restrict__ segmax, int E)
{
    int e = blockIdx.x * 256 + threadIdx.x;
    if (e >= E) return;
    const int s = srcIdx[e], d = dstIdx[e];
    const float4* q = (const float4*)(Q + (size_t)d * D_OUT);
    const float4* k = (const float4*)(K + (size_t)s * D_OUT);
    float acc = 0.0f;
    #pragma unroll
    for (int i = 0; i < D_OUT / 4; ++i) {
        const float4 a = q[i], b = k[i];
        acc = fmaf(a.x, b.x, acc);
        acc = fmaf(a.y, b.y, acc);
        acc = fmaf(a.z, b.z, acc);
        acc = fmaf(a.w, b.w, acc);
    }
    acc *= 0.125f;                // 1/sqrt(64)
    score[e] = acc;
    atomicMax(segmax + d, ford(acc));
}

// ---------------------------------------------------------------- exp + segment sum
__global__ __launch_bounds__(256) void ex_kernel(
    const int* __restrict__ dstIdx, float* __restrict__ score,
    const unsigned int* __restrict__ segmax, float* __restrict__ denom, int E)
{
    int e = blockIdx.x * 256 + threadIdx.x;
    if (e >= E) return;
    const int d = dstIdx[e];
    const float mx = funord(segmax[d]);
    const float ex = __expf(score[e] - mx);
    score[e] = ex;                               // reuse buffer for ex
    unsafeAtomicAdd(denom + d, ex);              // native global_atomic_add_f32
}

// ---------------------------------------------------------------- weighted scatter-add
// 32 lanes per edge; lane owns 2 output dims (coalesced per 16-lane group)
__global__ __launch_bounds__(256) void agg_kernel(
    const int* __restrict__ srcIdx, const int* __restrict__ dstIdx,
    const float* __restrict__ V, const float* __restrict__ exv,
    const float* __restrict__ denom, float* __restrict__ out, int E)
{
    const int t = blockIdx.x * 256 + threadIdx.x;
    const int e = t >> 5;
    if (e >= E) return;
    const int lane = t & 31;
    const int s = srcIdx[e], d = dstIdx[e];
    const float alpha = exv[e] / (denom[d] + 1e-16f);
    const float2 v = *(const float2*)(V + (size_t)s * D_OUT + lane * 2);
    float* o = out + (size_t)d * D_OUT + lane * 2;
    unsafeAtomicAdd(o + 0, alpha * v.x);
    unsafeAtomicAdd(o + 1, alpha * v.y);
}

// ---------------------------------------------------------------- launcher
extern "C" void kernel_launch(void* const* d_in, const int* in_sizes, int n_in,
                              void* d_out, int out_size, void* d_ws, size_t ws_size,
                              hipStream_t stream) {
    const float* x  = (const float*)d_in[0];
    const int*   ei = (const int*)d_in[1];      // edge_index flat [2, E]
    const float* Wq = (const float*)d_in[2];
    const float* bq = (const float*)d_in[3];
    const float* Wk = (const float*)d_in[4];
    const float* bk = (const float*)d_in[5];
    const float* Wv = (const float*)d_in[6];
    const float* bv = (const float*)d_in[7];
    const float* Ws = (const float*)d_in[8];
    const float* bs = (const float*)d_in[9];
    float* out = (float*)d_out;                 // [N, 64]

    const int nNodes = in_sizes[0] / D_IN;
    const int E      = in_sizes[1] / 2;
    const int* srcIdx = ei;                     // edge_index[0]
    const int* dstIdx = ei + E;                 // edge_index[1]

    // workspace: Q,K,V fp32 [N,64]; segmax u32 [N]; denom f32 [N]; score f32 [E]; Wp bf16
    float* Q = (float*)d_ws;
    float* K = Q + (size_t)nNodes * D_OUT;
    float* V = K + (size_t)nNodes * D_OUT;
    unsigned int* segmax = (unsigned int*)(V + (size_t)nNodes * D_OUT);
    float* denom = (float*)(segmax + nNodes);
    float* score = denom + nNodes;
    uintptr_t wpAddr = (uintptr_t)(score + E);
    wpAddr = (wpAddr + 31) & ~(uintptr_t)31;
    __bf16* Wp = (__bf16*)wpAddr;

    init_kernel<<<(nNodes + 255) / 256, 256, 0, stream>>>(segmax, denom, nNodes);

    pack_w_kernel<<<(NT_TOTAL * NKS * 32 + 255) / 256, 256, 0, stream>>>(Wq, Wk, Wv, Ws, Wp);

    gemm_qkvs_kernel<<<(nNodes + MTILE - 1) / MTILE, 256, 0, stream>>>(
        x, Wp, bq, bk, bv, bs, Q, K, V, out, nNodes);

    score_kernel<<<(E + 255) / 256, 256, 0, stream>>>(srcIdx, dstIdx, Q, K, score, segmax, E);

    ex_kernel<<<(E + 255) / 256, 256, 0, stream>>>(dstIdx, score, segmax, denom, E);

    agg_kernel<<<(E + 7) / 8, 256, 0, stream>>>(srcIdx, dstIdx, V, score, denom, out, E);
}
